// GCN_68942815035830
// MI455X (gfx1250) — matrix operand order repaired and checked by reading, not compiled, not run on hardware
//
#include <hip/hip_runtime.h>

typedef __attribute__((ext_vector_type(16))) _Float16 v16h;
typedef __attribute__((ext_vector_type(8)))  float    v8f;

#define GCN_N    100000
#define GCN_E    1600000
#define GCN_KIN  256
#define GCN_HID  128

// ---------------------------------------------------------------------------
// zero-fill (float4)
__global__ void gcn_zero4_kernel(float4* __restrict__ p, int n4) {
    int i = blockIdx.x * blockDim.x + threadIdx.x;
    if (i < n4) p[i] = make_float4(0.f, 0.f, 0.f, 0.f);
}

// degree count via float atomics: deg[idx[e]] += 1
__global__ void gcn_degree_kernel(const int* __restrict__ idx, float* __restrict__ deg, int E) {
    int i = blockIdx.x * blockDim.x + threadIdx.x;
    if (i < E) atomicAdd(&deg[idx[i]], 1.0f);
}

// in-place: deg -> rsqrt(max(deg,1))
__global__ void gcn_norm_kernel(float* __restrict__ deg, int n) {
    int i = blockIdx.x * blockDim.x + threadIdx.x;
    if (i < n) deg[i] = rsqrtf(fmaxf(deg[i], 1.0f));
}

// ---------------------------------------------------------------------------
// Pre-pack W[K x 128] fp32 -> f16 fragments in per-lane WMMA-B order.
// Fragment f = (kt*8 + ct)*32 + lane holds 16 halves: half j has
//   K = kt*32 + (lane>>4)*16 + j,  N = ct*16 + (lane&15)
__global__ void gcn_pack_w_kernel(const float* __restrict__ W, _Float16* __restrict__ Wp, int K) {
    int i = blockIdx.x * blockDim.x + threadIdx.x;
    if (i >= K * GCN_HID) return;
    int j    = i & 15;
    int lane = (i >> 4) & 31;
    int ct   = (i >> 9) & 7;
    int kt   = i >> 12;
    int kk   = kt * 32 + ((lane >> 4) << 4) + j;
    int nn   = (ct << 4) + (lane & 15);
    Wp[i] = (_Float16)W[(size_t)kk * GCN_HID + nn];
}

// ---------------------------------------------------------------------------
// WMMA GEMM: Y[M x 128] = (X[M x K] * rowscale[m]) @ W[K x 128] (+ bias[n])
// Block = 256 threads = 8 waves; wave w owns output tile (blockIdx.x*16, w*16).
// A tile staged fp32 via async global->LDS (CDNA5 path), converted to f16 at
// fragment build. B read as prepacked f16 fragments (coalesced b128 loads).
// Requires M%16==0, K%32==0 (holds: M=100000, K in {256,128}); EXEC all-ones
// at every WMMA.
template<bool SCALE>
__global__ void __launch_bounds__(256)
gcn_gemm_wmma_kernel(const float* __restrict__ X,
                     const float* __restrict__ rowscale,   // used iff SCALE
                     const _Float16* __restrict__ Wp,      // prepacked fragments
                     const float* __restrict__ bias,       // may be null
                     float* __restrict__ Y,
                     int K)
{
    // 16 rows x 32 floats, row stride 36 floats (144B: 16B-aligned chunks,
    // 36*r mod 64 distinct for r=0..15 -> conflict-free column reads)
    __shared__ float Asf[16 * 36];

    const int tid  = threadIdx.x;
    const int lane = tid & 31;
    const int wave = tid >> 5;
    const int rowBase = blockIdx.x * 16;
    const int mn  = lane & 15;        // A/C row within tile, B/C col within tile
    const int grp = lane >> 4;

    const float rs = SCALE ? rowscale[rowBase + mn] : 1.0f;

    // async-copy assignment: threads 0..127 each move one 16B chunk per K-step
    const int cr = tid >> 3;                       // row 0..15
    const int cc = (tid & 7) << 2;                 // col (floats) 0,4,...,28
    const unsigned ldsAddr =
        (unsigned)(size_t)(&Asf[0]) + (unsigned)(cr * 144 + ((tid & 7) << 4));

    v8f acc = {};
    const int nKT = K >> 5;
    for (int kt = 0; kt < nKT; ++kt) {
        const int k0 = kt << 5;
        if (tid < 128) {
            const float* g = X + (size_t)(rowBase + cr) * K + (k0 + cc);
            asm volatile("global_load_async_to_lds_b128 %0, %1, off"
                         :: "v"(ldsAddr), "v"(g) : "memory");
        }
        asm volatile("s_wait_asynccnt 0x0" ::: "memory");
        __syncthreads();

        // A fragment, contiguous k order matching prepacked B:
        //   K = grp*16 + j
        v16h a;
        #pragma unroll
        for (int j = 0; j < 16; ++j) {
            int vv = j >> 1;
            int kk = (grp << 4) + j; (void)vv;
            a[j] = (_Float16)(rs * Asf[mn * 36 + kk]);
        }

        // B fragment: prepacked, contiguous 32B per lane
        const v16h b = *(const v16h*)(Wp + ((size_t)((((kt << 3) + wave) << 5) + lane) << 4));

        acc = __builtin_amdgcn_wmma_f32_16x16x32_f16(
                  /*neg_a=*/false, a, /*neg_b=*/false, b,
                  /*c_mod=*/(short)0, acc, /*reuse_a=*/false, /*reuse_b=*/false);
        __syncthreads();
    }

    // C/D layout: VGPR i -> M = i + 8*grp, N = lane&15
    const int col = (wave << 4) + mn;
    const float bv = bias ? bias[col] : 0.0f;
    #pragma unroll
    for (int i = 0; i < 8; ++i) {
        int row = rowBase + i + (grp << 3);
        Y[(size_t)row * GCN_HID + col] = acc[i] + bv;
    }
}

// ---------------------------------------------------------------------------
// Edge scatter: Agg[dst[e]][:] += H[src[e]][:]. One thread per (edge, 4 feats).
__global__ void gcn_scatter_kernel(const float* __restrict__ H,
                                   const int* __restrict__ src,
                                   const int* __restrict__ dst,
                                   float* __restrict__ Agg, int E)
{
    long long idx   = (long long)blockIdx.x * blockDim.x + threadIdx.x;
    long long total = (long long)E * 32;
    if (idx >= total) return;
    int e = (int)(idx >> 5);
    int c = (int)(idx & 31) << 2;
    int s = src[e];
    int d = dst[e];
    const float4 v = *(const float4*)(H + (size_t)s * GCN_HID + c);
    float* p = Agg + (size_t)d * GCN_HID + c;
    atomicAdd(p + 0, v.x);
    atomicAdd(p + 1, v.y);
    atomicAdd(p + 2, v.z);
    atomicAdd(p + 3, v.w);
}

// out[i,:] = relu(Agg[i,:] * nin[i] + b[:]), float4-vectorized; in-place safe
__global__ void gcn_finalize_kernel(const float4* __restrict__ Agg,
                                    const float* __restrict__ nin,
                                    const float4* __restrict__ b4,
                                    float4* __restrict__ Out, int n4)
{
    int i = blockIdx.x * blockDim.x + threadIdx.x;
    if (i < n4) {
        int row = i >> 5;            // 32 float4 per 128-wide row
        int c4  = i & 31;
        float s = nin[row];
        float4 v = Agg[i];
        float4 bb = b4[c4];
        float4 o;
        o.x = fmaxf(v.x * s + bb.x, 0.f);
        o.y = fmaxf(v.y * s + bb.y, 0.f);
        o.z = fmaxf(v.z * s + bb.z, 0.f);
        o.w = fmaxf(v.w * s + bb.w, 0.f);
        Out[i] = o;
    }
}

// ---------------------------------------------------------------------------
extern "C" void kernel_launch(void* const* d_in, const int* in_sizes, int n_in,
                              void* d_out, int out_size, void* d_ws, size_t ws_size,
                              hipStream_t stream) {
    const float* x   = (const float*)d_in[0];
    const float* We  = (const float*)d_in[1];
    const float* be  = (const float*)d_in[2];
    const float* W1  = (const float*)d_in[3];
    const float* b1  = (const float*)d_in[4];
    const float* W2  = (const float*)d_in[5];
    const float* b2  = (const float*)d_in[6];
    const int* src0  = (const int*)d_in[7];
    const int* dst0  = (const int*)d_in[8];
    const int* src1  = (const int*)d_in[9];
    const int* dst1  = (const int*)d_in[10];
    float* out = (float*)d_out;

    const int Nn = GCN_N, Ee = GCN_E;
    const int feat_total = Nn * GCN_HID;          // 12.8M floats

    // workspace layout: 4 norm vectors | H0 | H1 | packed weights (f16)
    float* nOut0 = (float*)d_ws;                   // N
    float* nIn0  = nOut0 + Nn;
    float* nOut1 = nIn0  + Nn;
    float* nIn1  = nOut1 + Nn;
    float* H0    = nIn1  + Nn;                     // N x 128
    float* H1    = H0 + (size_t)feat_total;        // N x 128
    _Float16* WeP = (_Float16*)(H1 + (size_t)feat_total); // 256*128 halves
    _Float16* W1P = WeP + GCN_KIN * GCN_HID;               // 128*128 halves
    _Float16* W2P = W1P + GCN_HID * GCN_HID;

    const int TB = 256;
    const int gridRows = Nn / 16;                                  // 6250
    const int gridE    = (Ee + TB - 1) / TB;
    const int gridF4   = (feat_total / 4 + TB - 1) / TB;
    const int gridN4   = (4 * Nn + TB - 1) / TB;
    const long long scatterTot = (long long)Ee * 32;
    const int gridS    = (int)((scatterTot + TB - 1) / TB);

    // --- pack weights into WMMA-B fragment order (tiny, once per call) ---
    gcn_pack_w_kernel<<<(GCN_KIN * GCN_HID) / TB, TB, 0, stream>>>(We, WeP, GCN_KIN);
    gcn_pack_w_kernel<<<(GCN_HID * GCN_HID) / TB, TB, 0, stream>>>(W1, W1P, GCN_HID);
    gcn_pack_w_kernel<<<(GCN_HID * GCN_HID) / TB, TB, 0, stream>>>(W2, W2P, GCN_HID);

    // --- degrees -> norms (all four lists) ---
    gcn_zero4_kernel<<<(Nn + TB - 1) / TB, TB, 0, stream>>>((float4*)nOut0, Nn); // 4N floats = N float4
    gcn_degree_kernel<<<gridE, TB, 0, stream>>>(src0, nOut0, Ee);
    gcn_degree_kernel<<<gridE, TB, 0, stream>>>(dst0, nIn0,  Ee);
    gcn_degree_kernel<<<gridE, TB, 0, stream>>>(src1, nOut1, Ee);
    gcn_degree_kernel<<<gridE, TB, 0, stream>>>(dst1, nIn1,  Ee);
    gcn_norm_kernel<<<gridN4, TB, 0, stream>>>(nOut0, 4 * Nn);

    // --- embed: H0 = x @ We + be ---
    gcn_gemm_wmma_kernel<false><<<gridRows, TB, 0, stream>>>(x, nullptr, WeP, be, H0, GCN_KIN);

    // --- layer 1 ---
    gcn_gemm_wmma_kernel<true><<<gridRows, TB, 0, stream>>>(H0, nOut0, W1P, nullptr, H1, GCN_HID);
    gcn_zero4_kernel<<<gridF4, TB, 0, stream>>>((float4*)out, feat_total / 4);
    gcn_scatter_kernel<<<gridS, TB, 0, stream>>>(H1, src0, dst0, out, Ee);
    gcn_finalize_kernel<<<gridF4, TB, 0, stream>>>((const float4*)out, nIn0,
                                                   (const float4*)b1, (float4*)H0, feat_total / 4);

    // --- layer 2 ---
    gcn_gemm_wmma_kernel<true><<<gridRows, TB, 0, stream>>>(H0, nOut1, W2P, nullptr, H1, GCN_HID);
    gcn_zero4_kernel<<<gridF4, TB, 0, stream>>>((float4*)out, feat_total / 4);
    gcn_scatter_kernel<<<gridS, TB, 0, stream>>>(H1, src1, dst1, out, Ee);
    gcn_finalize_kernel<<<gridF4, TB, 0, stream>>>((const float4*)out, nIn1,
                                                   (const float4*)b2, (float4*)out, feat_total / 4);
}
